// LSHLinearStrided_61529701483102
// MI455X (gfx1250) — matrix operand order repaired and checked by reading, not compiled, not run on hardware
//
#include <hip/hip_runtime.h>
#include <stdint.h>

// ---------------------------------------------------------------------------
// LSH masked linear for MI455X (gfx1250, wave32).
//   out[m,n] = collide(m,n) ? (x[m,:] . W[n,:] + b[n]) : 0
// GEMM in bf16 WMMA (f32 accum), double-buffered LDS staging with hardware
// v_cvt_pk_bf16_f32 converts; codes + mask in fp32/int VALU; NT output stores.
// ---------------------------------------------------------------------------

typedef __attribute__((ext_vector_type(16))) __bf16 v16bf;
typedef __attribute__((ext_vector_type(2)))  __bf16 v2bf;
typedef __attribute__((ext_vector_type(8)))  float  v8f;

constexpr int D_DIM    = 1024;   // feature dim (fixed by reference)
constexpr int N_TABLES = 8;      // T
constexpr int H_BITS   = 6;      // bits per table

constexpr int BM = 128;          // M tile per block
constexpr int BN = 128;          // N tile per block
constexpr int BK = 32;           // K step (one bf16 WMMA K)
constexpr int NTILES = D_DIM / BK;                 // 32 K-tiles
constexpr int LDS_STRIDE = 20;   // dwords per LDS row: 16 data (32 bf16) + 4 pad
constexpr int A_DWORDS = BM * LDS_STRIDE;          // 2560 dwords
constexpr int BUF_DWORDS = (BM + BN) * LDS_STRIDE; // A tile + B tile per buffer

// ------------------------------ codes kernel -------------------------------
// One 64-thread block per row (token rows first, then neuron rows).
__global__ __launch_bounds__(64) void lsh_codes_kernel(
    const float* __restrict__ x, const float* __restrict__ W,
    const float* __restrict__ proj, uint8_t* __restrict__ codes,
    int M, int O)
{
    __shared__ float vbuf[D_DIM];
    __shared__ int   sbits[N_TABLES * H_BITS];

    const int e   = blockIdx.x;
    const int tid = threadIdx.x;
    const float* v = (e < M) ? (x + (size_t)e * D_DIM)
                             : (W + (size_t)(e - M) * D_DIM);

    // Stage the row into LDS: 64 threads x 4 float4 = 1024 floats.
    const float4* v4  = (const float4*)v;
    float4*       vb4 = (float4*)vbuf;
#pragma unroll
    for (int i = 0; i < 4; ++i) vb4[tid + 64 * i] = v4[tid + 64 * i];
    __syncthreads();

    if (tid < N_TABLES * H_BITS) {
        const float4* p4  = (const float4*)(proj + (size_t)tid * D_DIM);
        const float4* vv4 = (const float4*)vbuf;
        float acc = 0.f;
#pragma unroll 4
        for (int i = 0; i < D_DIM / 4; ++i) {
            float4 p = p4[i], q = vv4[i];
            acc += p.x * q.x + p.y * q.y + p.z * q.z + p.w * q.w;
        }
        sbits[tid] = (acc > 0.f) ? 1 : 0;   // reference uses strict '> 0'
    }
    __syncthreads();

    if (tid < N_TABLES) {
        int c = 0;
#pragma unroll
        for (int h = 0; h < H_BITS; ++h) c |= sbits[tid * H_BITS + h] << h;
        codes[(size_t)e * N_TABLES + tid] = (uint8_t)c;
    }
}

// ------------- pack f32 pair -> bf16 pair (one v_cvt_pk_bf16_f32) ----------
__device__ inline unsigned pack2_bf16(float a, float b)
{
    v2bf v;
    v[0] = (__bf16)a;
    v[1] = (__bf16)b;
    return __builtin_bit_cast(unsigned, v);
}

__device__ inline uint4 pack8_bf16(float4 p, float4 q)
{
    uint4 r;
    r.x = pack2_bf16(p.x, p.y);
    r.y = pack2_bf16(p.z, p.w);
    r.z = pack2_bf16(q.x, q.y);
    r.w = pack2_bf16(q.z, q.w);
    return r;
}

// ------------------------------ GEMM kernel --------------------------------
__global__ __launch_bounds__(256) void lsh_linear_wmma_kernel(
    const float* __restrict__ x, const float* __restrict__ W,
    const float* __restrict__ bias, const uint8_t* __restrict__ codes,
    float* __restrict__ out, int M, int O)
{
    // Ping-pong buffers: [buf][A tile | B tile], 2 x 20 KB = 40 KB.
    __shared__ unsigned lds[2][BUF_DWORDS];

    const int tid  = threadIdx.x;
    const int lane = tid & 31;
    const int wave = tid >> 5;        // 0..7
    const int wm   = wave & 3;        // 32-row slice within block tile
    const int wn   = wave >> 2;       // 64-col slice within block tile
    const int m_blk = blockIdx.y * BM;
    const int n_blk = blockIdx.x * BN;

    // Staging map: each thread owns (row, 16-float segment) of both tiles.
    const int srow = tid >> 1;        // 0..127
    const int sseg = tid & 1;         // 0/1
    const float* gA = x + (size_t)(m_blk + srow) * D_DIM + sseg * 16;
    const float* gB = W + (size_t)(n_blk + srow) * D_DIM + sseg * 16;
    const int sOffA = srow * LDS_STRIDE + sseg * 8;            // dwords
    const int sOffB = A_DWORDS + srow * LDS_STRIDE + sseg * 8; // dwords

    v8f acc[2][4];
#pragma unroll
    for (int mt = 0; mt < 2; ++mt)
#pragma unroll
        for (int nt = 0; nt < 4; ++nt)
            acc[mt][nt] = (v8f){0.f,0.f,0.f,0.f,0.f,0.f,0.f,0.f};

    // ---- prologue: stage K tile 0 into buffer 0
    {
        float4 a0 = *(const float4*)(gA +  0);
        float4 a1 = *(const float4*)(gA +  4);
        float4 a2 = *(const float4*)(gA +  8);
        float4 a3 = *(const float4*)(gA + 12);
        float4 b0 = *(const float4*)(gB +  0);
        float4 b1 = *(const float4*)(gB +  4);
        float4 b2 = *(const float4*)(gB +  8);
        float4 b3 = *(const float4*)(gB + 12);
        __builtin_prefetch(gA + BK, 0, 3);   // WGP-scope hint for next tile
        __builtin_prefetch(gB + BK, 0, 3);
        *(uint4*)&lds[0][sOffA + 0] = pack8_bf16(a0, a1);
        *(uint4*)&lds[0][sOffA + 4] = pack8_bf16(a2, a3);
        *(uint4*)&lds[0][sOffB + 0] = pack8_bf16(b0, b1);
        *(uint4*)&lds[0][sOffB + 4] = pack8_bf16(b2, b3);
    }
    __syncthreads();

    // Fragment addressing (ISA 7.12.2 bf16 layouts).
    const int r16   = lane & 15;
    const int akoff = (lane >> 4) * 4;   // A: lanes16-31 start at K=8 (4 dw)
    const int bkoff = (lane >> 4) * 4;   // B: lanes16-31 start at K=8 (4 dw)

    // ---- steady state: NTILES-1 iterations stage the next tile
    for (int t = 0; t < NTILES - 1; ++t) {
        const int cur = t & 1;
        const int nxt = cur ^ 1;
        const int kn  = (t + 1) * BK;

        // next tile's global loads first (overlap with this tile's WMMAs)
        float4 a0 = *(const float4*)(gA + kn +  0);
        float4 a1 = *(const float4*)(gA + kn +  4);
        float4 a2 = *(const float4*)(gA + kn +  8);
        float4 a3 = *(const float4*)(gA + kn + 12);
        float4 b0 = *(const float4*)(gB + kn +  0);
        float4 b1 = *(const float4*)(gB + kn +  4);
        float4 b2 = *(const float4*)(gB + kn +  8);
        float4 b3 = *(const float4*)(gB + kn + 12);
        if (t + 2 < NTILES) {                // next-next tile into WGP caches
            __builtin_prefetch(gA + kn + BK, 0, 3);
            __builtin_prefetch(gB + kn + BK, 0, 3);
        }

        // fragments from current buffer
        const unsigned* curA = &lds[cur][0];
        const unsigned* curB = &lds[cur][A_DWORDS];
        v16bf afrag[2], bfrag[4];
#pragma unroll
        for (int mt = 0; mt < 2; ++mt) {
            const unsigned* p = curA + (wm * 32 + mt * 16 + r16) * LDS_STRIDE + akoff;
            union { uint4 q[2]; v16bf v; } u;
            u.q[0] = *(const uint4*)(p);        // K 0-7   (or 8-15)
            u.q[1] = *(const uint4*)(p + 8);    // K 16-23 (or 24-31)
            afrag[mt] = u.v;
        }
#pragma unroll
        for (int nt = 0; nt < 4; ++nt) {
            const unsigned* p = curB + (wn * 64 + nt * 16 + r16) * LDS_STRIDE + bkoff;
            union { uint4 q[2]; v16bf v; } u;
            u.q[0] = *(const uint4*)(p);        // K 0-7   (or 8-15)
            u.q[1] = *(const uint4*)(p + 8);    // K 16-23 (or 24-31)
            bfrag[nt] = u.v;
        }

#pragma unroll
        for (int mt = 0; mt < 2; ++mt)
#pragma unroll
            for (int nt = 0; nt < 4; ++nt)
                acc[mt][nt] = __builtin_amdgcn_wmma_f32_16x16x32_bf16(
                    false, afrag[mt], false, bfrag[nt],
                    (short)0, acc[mt][nt], false, false);

        // convert + store next tile into the other buffer
        *(uint4*)&lds[nxt][sOffA + 0] = pack8_bf16(a0, a1);
        *(uint4*)&lds[nxt][sOffA + 4] = pack8_bf16(a2, a3);
        *(uint4*)&lds[nxt][sOffB + 0] = pack8_bf16(b0, b1);
        *(uint4*)&lds[nxt][sOffB + 4] = pack8_bf16(b2, b3);

        // One barrier per K step: protects writes to lds[nxt] (read next
        // iter) AND reads of lds[cur] (overwritten next iter).
        __syncthreads();
    }

    // ---- peeled final tile: fragments + WMMA only
    {
        const int cur = (NTILES - 1) & 1;
        const unsigned* curA = &lds[cur][0];
        const unsigned* curB = &lds[cur][A_DWORDS];
        v16bf afrag[2], bfrag[4];
#pragma unroll
        for (int mt = 0; mt < 2; ++mt) {
            const unsigned* p = curA + (wm * 32 + mt * 16 + r16) * LDS_STRIDE + akoff;
            union { uint4 q[2]; v16bf v; } u;
            u.q[0] = *(const uint4*)(p);
            u.q[1] = *(const uint4*)(p + 8);
            afrag[mt] = u.v;
        }
#pragma unroll
        for (int nt = 0; nt < 4; ++nt) {
            const unsigned* p = curB + (wn * 64 + nt * 16 + r16) * LDS_STRIDE + bkoff;
            union { uint4 q[2]; v16bf v; } u;
            u.q[0] = *(const uint4*)(p);
            u.q[1] = *(const uint4*)(p + 8);
            bfrag[nt] = u.v;
        }
#pragma unroll
        for (int mt = 0; mt < 2; ++mt)
#pragma unroll
            for (int nt = 0; nt < 4; ++nt)
                acc[mt][nt] = __builtin_amdgcn_wmma_f32_16x16x32_bf16(
                    false, afrag[mt], false, bfrag[nt],
                    (short)0, acc[mt][nt], false, false);
    }

    // ---- epilogue: LSH collision mask + bias + NT store
    const unsigned long long* cx = (const unsigned long long*)codes;      // [M]
    const unsigned long long* cw = cx + M;                                // [O]
    const int col   = lane & 15;
    const int rhalf = (lane >> 4) * 8;   // C/D layout: lanes16-31 hold M=r+8
#pragma unroll
    for (int nt = 0; nt < 4; ++nt) {
        const int n = n_blk + wn * 64 + nt * 16 + col;
        const unsigned long long wcode = cw[n];
        const float bn = bias[n];
#pragma unroll
        for (int mt = 0; mt < 2; ++mt) {
#pragma unroll
            for (int r = 0; r < 8; ++r) {
                const int m = m_blk + wm * 32 + mt * 16 + r + rhalf;
                unsigned long long z = cx[m] ^ wcode;
                // collide iff any of the 8 per-table byte codes match
                bool hit = ((z - 0x0101010101010101ULL) & ~z &
                            0x8080808080808080ULL) != 0ULL;
                float val = hit ? (acc[mt][nt][r] + bn) : 0.0f;
                // output is write-once, never re-read: keep it out of L2
                __builtin_nontemporal_store(val, &out[(size_t)m * O + n]);
            }
        }
    }
}

// ------------------------------- launcher ----------------------------------
extern "C" void kernel_launch(void* const* d_in, const int* in_sizes, int n_in,
                              void* d_out, int out_size, void* d_ws, size_t ws_size,
                              hipStream_t stream)
{
    const float* x    = (const float*)d_in[0];   // [B,S,D]
    const float* W    = (const float*)d_in[1];   // [O,D]
    const float* b    = (const float*)d_in[2];   // [O]
    const float* proj = (const float*)d_in[3];   // [T,Hb,D]
    float* out        = (float*)d_out;           // [B,S,O]

    const int O = in_sizes[2];                   // 4096
    const int M = in_sizes[0] / D_DIM;           // B*S = 8192

    uint8_t* codes = (uint8_t*)d_ws;             // (M+O)*8 bytes = 96 KB

    lsh_codes_kernel<<<M + O, 64, 0, stream>>>(x, W, proj, codes, M, O);

    dim3 grid(O / BN, M / BM);                   // 32 x 64 blocks
    lsh_linear_wmma_kernel<<<grid, 256, 0, stream>>>(x, W, b, codes, out, M, O);
}
